// SegPatchLoss_62208306315429
// MI455X (gfx1250) — compile-verified, matrix operand order checked by reading, not run on hardware
//
#include <hip/hip_runtime.h>
#include <hip/hip_bf16.h>

typedef __attribute__((ext_vector_type(16))) _Float16 v16h;
typedef __attribute__((ext_vector_type(8)))  float    v8f;

#define TOPN 5
#define BIGF 3.0e38f

// ---------------------------------------------------------------------------
// Distance kernel: for one target patch (blockIdx.y) compute the summed-L1
// distance from the target patch to 16 source-patch positions per wave,
// using V_WMMA_F32_16X16X32_F16 with an all-ones B matrix as a 16x32
// reduction engine (D[m][n] = sum_k A[m][k]).
//
// Because B is all ones, the per-lane slot->k mapping is a free bijection:
// lane-half 0 supplies k in [0, K/2), lane-half 1 supplies k in [K/2, K).
// K/2 is divisible by PS*PS, so the half offset folds into the channel base
// and every element offset is a compile-time constant off one base pointer.
//
// NOTE: all arithmetic around the abs is kept strictly in f16. Any f32 fabs
// (even one reachable via the fabs(fptrunc x) -> fptrunc(fabs x) hoist) gets
// SLP-vectorized to fabs.v2f32, which crashes this gfx1250 backend's ISel.
// With s and t both pre-converted to f16, the fabs operand is an f16 fsub,
// so only fabs.f16/v2f16 (legal and-mask lowering) can be formed.
// ---------------------------------------------------------------------------
template<int C, int H, int W, int PS>
__global__ __launch_bounds__(256) void dist_kernel(
    const float* __restrict__ src, const float* __restrict__ tgt,
    const int* __restrict__ pos, float* __restrict__ dist)
{
  constexpr int HW   = H * W;
  constexpr int Hm   = H - PS, Wm = W - PS;
  constexpr int P    = Hm * Wm;              // number of source patches
  constexpr int K    = C * PS * PS;          // patch length
  constexpr int KC   = K / 32;               // 32-wide k-chunks per patch
  constexpr int KH   = K / 2;                // k range per lane-half
  constexpr int CH   = KH / (PS * PS);       // channel offset of upper half
  constexpr int PADP = ((P + 15) / 16) * 16; // padded distance row length

  const int t  = blockIdx.y;
  const int th = pos[2 * t + 0];
  const int tw = pos[2 * t + 1];

  // Stage the target patch in LDS as f16, k = c*PS*PS + i*PS + j.
  __shared__ _Float16 tpatch[K];
  for (int k = threadIdx.x; k < K; k += blockDim.x) {
    const int c = k / (PS * PS);
    const int r = k % (PS * PS);
    const int i = r / PS, j = r % PS;
    tpatch[k] = (_Float16)tgt[c * HW + (th + i) * W + (tw + j)];
  }
  __syncthreads();

  const int wave = threadIdx.x >> 5;
  const int lane = threadIdx.x & 31;
  const int p0   = (blockIdx.x * 8 + wave) * 16;   // 8 waves per block
  if (p0 >= PADP) return;                          // wave-uniform exit

  const int half = lane >> 4;     // which k-half this lane feeds
  const int m    = lane & 15;     // A-matrix row owned by this lane
  int p = p0 + m;
  if (p >= P) p = P - 1;          // clamp; padded rows overwritten with +INF
  const int h = p / Wm;
  const int w = p - h * Wm;

  const float* __restrict__ base = src + (h * W + w) + half * (CH * HW);
  const _Float16* __restrict__ tb = tpatch + half * KH;

  v16h ones;
  #pragma unroll
  for (int e = 0; e < 16; ++e) ones[e] = (_Float16)1.0f;

  v8f acc = {};
  #pragma unroll
  for (int kc = 0; kc < KC; ++kc) {
    v16h a;
    #pragma unroll
    for (int e = 0; e < 16; ++e) {
      const int q  = kc * 16 + e;            // compile-time per-lane k index
      const int cq = q / (PS * PS);
      const int r  = q % (PS * PS);
      const int i  = r / PS, j = r % PS;
      const float s = base[cq * HW + i * W + j];   // immediate-offset load
      const _Float16 hd = (_Float16)s - tb[q];     // f16 subtract
      a[e] = __builtin_fabsf16(hd);                // f16 abs only
    }
    // D = A x Ones + C : each row of D is the partial L1 distance of patch m
    acc = __builtin_amdgcn_wmma_f32_16x16x32_f16(
        /*neg_a=*/false, a, /*neg_b=*/false, ones,
        /*c_mod=*/(short)0, acc, /*reuse_a=*/false, /*reuse_b=*/false);
  }

  // C/D layout: lane 0 (N=0) holds rows 0..7 in acc[0..7]; lane 16 rows 8..15.
  if (m == 0) {
    float* dptr = dist + (size_t)t * PADP;
    #pragma unroll
    for (int q = 0; q < 8; ++q) {
      const int pp = p0 + half * 8 + q;
      dptr[pp] = (pp < P) ? acc[q] : BIGF;   // pad with +INF sentinels
    }
  }
}

// ---------------------------------------------------------------------------
// Per-target top-5 + loss accumulation. One block per target.
// ---------------------------------------------------------------------------
__global__ __launch_bounds__(256) void topk_kernel(
    const float* __restrict__ dist, int padStride, float invScale,
    float* __restrict__ out)
{
  const float* d = dist + (size_t)blockIdx.x * padStride;

  float best[TOPN];
  #pragma unroll
  for (int q = 0; q < TOPN; ++q) best[q] = BIGF;

  for (int i = threadIdx.x; i < padStride; i += 256) {
    const float v = d[i];
    if (v < best[TOPN - 1]) {
      best[TOPN - 1] = v;
      #pragma unroll
      for (int q = TOPN - 1; q > 0; --q)
        if (best[q] < best[q - 1]) {
          float tmp = best[q]; best[q] = best[q - 1]; best[q - 1] = tmp;
        }
    }
  }

  __shared__ float cand[256 * TOPN];
  #pragma unroll
  for (int q = 0; q < TOPN; ++q) cand[threadIdx.x * TOPN + q] = best[q];
  __syncthreads();

  if (threadIdx.x == 0) {
    float b2[TOPN];
    #pragma unroll
    for (int q = 0; q < TOPN; ++q) b2[q] = BIGF;
    for (int i = 0; i < 256 * TOPN; ++i) {
      const float v = cand[i];
      if (v < b2[TOPN - 1]) {
        b2[TOPN - 1] = v;
        for (int q = TOPN - 1; q > 0; --q)
          if (b2[q] < b2[q - 1]) {
            float tmp = b2[q]; b2[q] = b2[q - 1]; b2[q - 1] = tmp;
          }
      }
    }
    float s = 0.f;
    #pragma unroll
    for (int q = 0; q < TOPN; ++q) s += b2[q];
    atomicAdd(out, s * invScale);
  }
}

__global__ void zero_kernel(float* __restrict__ out) {
  if (threadIdx.x == 0 && blockIdx.x == 0) out[0] = 0.0f;
}

// ---------------------------------------------------------------------------
extern "C" void kernel_launch(void* const* d_in, const int* in_sizes, int n_in,
                              void* d_out, int out_size, void* d_ws, size_t ws_size,
                              hipStream_t stream) {
  (void)n_in; (void)out_size; (void)ws_size;

  const float* src0 = (const float*)d_in[0];   // [1,64,256,256]
  const float* tgt0 = (const float*)d_in[1];   // [1,64,256,256]
  const float* src1 = (const float*)d_in[2];   // [1,128,128,128]
  const float* tgt1 = (const float*)d_in[3];   // [1,128,128,128]
  const int*   pos0 = (const int*)d_in[4];     // [10,2]
  const int*   pos1 = (const int*)d_in[5];     // [5,2]
  float*       out  = (float*)d_out;

  const int nT0 = in_sizes[4] / 2;             // 10
  const int nT1 = in_sizes[5] / 2;             // 5

  constexpr int P0   = 253 * 253;              // 64009
  constexpr int P1   = 127 * 127;              // 16129
  constexpr int PAD0 = ((P0 + 15) / 16) * 16;  // 64016
  constexpr int PAD1 = ((P1 + 15) / 16) * 16;  // 16144
  constexpr int G0   = (P0 + 15) / 16;         // wave groups, scale 0
  constexpr int G1   = (P1 + 15) / 16;

  float* dist0 = (float*)d_ws;
  float* dist1 = dist0 + (size_t)10 * PAD0;    // total ws use ~2.9 MB

  zero_kernel<<<1, 32, 0, stream>>>(out);

  dist_kernel<64, 256, 256, 3>
      <<<dim3((G0 + 7) / 8, nT0), 256, 0, stream>>>(src0, tgt0, pos0, dist0);
  dist_kernel<128, 128, 128, 1>
      <<<dim3((G1 + 7) / 8, nT1), 256, 0, stream>>>(src1, tgt1, pos1, dist1);

  // loss_t = sum(top5)/(K*5); total = sum_t / PATCH_NUM
  topk_kernel<<<nT0, 256, 0, stream>>>(dist0, PAD0, 1.0f / (576.f * 5.f * 10.f), out);
  topk_kernel<<<nT1, 256, 0, stream>>>(dist1, PAD1, 1.0f / (128.f * 5.f * 10.f), out);
}
